// MultiHeadAttention_35905926595506
// MI455X (gfx1250) — matrix operand order repaired and checked by reading, not compiled, not run on hardware
//
#include <hip/hip_runtime.h>

// ---------------------------------------------------------------------------
// MHA for MI455X (gfx1250): bf16 WMMA everywhere, fp32 accumulate.
//   1) convert x, Wq, Wk, Wv, Wo fp32 -> bf16 (workspace)
//   2) GEMM (v_wmma_f32_16x16x32_bf16, double-buffered LDS slabs staged with
//      GLOBAL_LOAD_ASYNC_TO_LDS_B128 when available): Q,K -> [B,NH,S,HD],
//      V -> [B,NH,HD,S] (transposed)
//   3) flash attention (causal), per-wave 16 q-rows, online softmax, bf16 P via LDS
//   4) GEMM: ctx @ Wo^T + bo -> fp32 d_out
// ---------------------------------------------------------------------------

typedef __bf16 bf16_t;
typedef __attribute__((ext_vector_type(4)))  bf16_t v4bf;
typedef __attribute__((ext_vector_type(8)))  bf16_t v8bf;
typedef __attribute__((ext_vector_type(16))) bf16_t v16bf;
typedef __attribute__((ext_vector_type(8)))  float  v8f;
typedef __attribute__((ext_vector_type(4)))  int    v4i;

typedef __attribute__((address_space(1))) v4i v4i_gl;   // global int4
typedef __attribute__((address_space(3))) v4i v4i_lds;  // LDS int4

#if __has_builtin(__builtin_amdgcn_global_load_async_to_lds_b128) && \
    __has_builtin(__builtin_amdgcn_s_wait_asynccnt)
#define USE_ASYNC_LDS 1
#else
#define USE_ASYNC_LDS 0
#endif

#define B_  2
#define S_  2048
#define H_  2048
#define NH_ 16
#define HD_ 128
#define M_  (B_*S_)

static __device__ __forceinline__ v16bf bf_cat(v8bf lo, v8bf hi) {
  return __builtin_shufflevector(lo, hi, 0,1,2,3,4,5,6,7,8,9,10,11,12,13,14,15);
}

// ---------------- fp32 -> bf16 conversion (vectorized) ----------------------
__global__ void __launch_bounds__(256)
cvt_f32_bf16_v4(const float4* __restrict__ src, v4bf* __restrict__ dst, int n4) {
  int i = blockIdx.x * blockDim.x + threadIdx.x;
  if (i >= n4) return;
  float4 f = src[i];
  v4bf o;
  o[0] = (bf16_t)f.x; o[1] = (bf16_t)f.y; o[2] = (bf16_t)f.z; o[3] = (bf16_t)f.w;
  dst[i] = o;
}

// ---------------- GEMM: C = A[M,K] @ W[N,K]^T + bias ------------------------
// MODE 0: bf16 out, split-head layout   out[((b*NH+h)*S + s)*HD + d]
// MODE 1: bf16 out, transposed-V layout out[((b*NH+h)*HD + d)*S + s]
// MODE 2: f32 out, row-major            out[m*N + n]
// Block: 128x128 C tile, 256 threads = 8 waves arranged 4(M) x 2(N).
// K slabs of 32 double-buffered in LDS; slab t+1 DMA'd (async-to-LDS) while
// slab t feeds 8 WMMAs per wave.
template<int MODE>
__global__ void __launch_bounds__(256)
gemm_bf16_nt(const bf16_t* __restrict__ A, const bf16_t* __restrict__ W,
             const float* __restrict__ bias, void* __restrict__ out,
             int N, int K) {
  __shared__ __align__(64) bf16_t As[2][128 * 32];
  __shared__ __align__(64) bf16_t Bs[2][128 * 32];

  const int tid   = threadIdx.x;
  const int lane  = tid & 31;
  const int wave  = tid >> 5;
  const int wm    = wave & 3;   // M group: wm*32
  const int wn    = wave >> 2;  // N group: wn*64
  const int mbase = blockIdx.y * 128;
  const int nbase = blockIdx.x * 128;
  const int l16   = lane & 15;
  const int half  = (lane < 16) ? 0 : 1;

  // this thread's two 16B staging chunks (chunk -> row, 16B column)
  const int row0 = (tid * 2) >> 2;
  const int col0 = ((tid * 2) & 3) * 8;
  const int row1 = (tid * 2 + 1) >> 2;
  const int col1 = ((tid * 2 + 1) & 3) * 8;

  auto stage = [&](int k0, int buf) {
#if USE_ASYNC_LDS
    __builtin_amdgcn_global_load_async_to_lds_b128(
        (v4i_gl*)&A[(size_t)(mbase + row0) * K + k0 + col0],
        (v4i_lds*)&As[buf][row0 * 32 + col0], 0, 0);
    __builtin_amdgcn_global_load_async_to_lds_b128(
        (v4i_gl*)&A[(size_t)(mbase + row1) * K + k0 + col1],
        (v4i_lds*)&As[buf][row1 * 32 + col1], 0, 0);
    __builtin_amdgcn_global_load_async_to_lds_b128(
        (v4i_gl*)&W[(size_t)(nbase + row0) * K + k0 + col0],
        (v4i_lds*)&Bs[buf][row0 * 32 + col0], 0, 0);
    __builtin_amdgcn_global_load_async_to_lds_b128(
        (v4i_gl*)&W[(size_t)(nbase + row1) * K + k0 + col1],
        (v4i_lds*)&Bs[buf][row1 * 32 + col1], 0, 0);
#else
    *(v8bf*)&As[buf][row0 * 32 + col0] = *(const v8bf*)&A[(size_t)(mbase + row0) * K + k0 + col0];
    *(v8bf*)&As[buf][row1 * 32 + col1] = *(const v8bf*)&A[(size_t)(mbase + row1) * K + k0 + col1];
    *(v8bf*)&Bs[buf][row0 * 32 + col0] = *(const v8bf*)&W[(size_t)(nbase + row0) * K + k0 + col0];
    *(v8bf*)&Bs[buf][row1 * 32 + col1] = *(const v8bf*)&W[(size_t)(nbase + row1) * K + k0 + col1];
#endif
  };

  v8f acc[2][4] = {};

  stage(0, 0);
#if USE_ASYNC_LDS
  __builtin_amdgcn_s_wait_asynccnt(0);
#endif

  int cur = 0;
  for (int k0 = 0; k0 < K; k0 += 32) {
    __syncthreads();                       // prev iter's LDS reads done everywhere
    if (k0 + 32 < K) stage(k0 + 32, cur ^ 1);  // overlap DMA with WMMAs below

    // A fragments: lane<16 -> K {0..7,16..23}; lane>=16 -> K {8..15,24..31}
    v16bf af[2];
#pragma unroll
    for (int r = 0; r < 2; ++r) {
      const bf16_t* p = &As[cur][(wm * 32 + r * 16 + l16) * 32 + half * 8];
      af[r] = bf_cat(*(const v8bf*)p, *(const v8bf*)(p + 16));
    }
    // B fragments: lane holds col N=l16; lane<16 -> K {0..7,16..23}; lane>=16 -> K {8..15,24..31}
    v16bf bfm[4];
#pragma unroll
    for (int j = 0; j < 4; ++j) {
      const bf16_t* p = &Bs[cur][(wn * 64 + j * 16 + l16) * 32 + half * 8];
      bfm[j] = bf_cat(*(const v8bf*)p, *(const v8bf*)(p + 16));
    }
#pragma unroll
    for (int r = 0; r < 2; ++r)
#pragma unroll
      for (int j = 0; j < 4; ++j)
        acc[r][j] = __builtin_amdgcn_wmma_f32_16x16x32_bf16(
            false, af[r], false, bfm[j], (short)0, acc[r][j], false, false);

#if USE_ASYNC_LDS
    __builtin_amdgcn_s_wait_asynccnt(0);   // prefetched slab has landed
#endif
    cur ^= 1;
  }

  // epilogue: bias + store (C layout: row = v + (half?8:0), col = l16)
#pragma unroll
  for (int r = 0; r < 2; ++r) {
    int mrow0 = mbase + wm * 32 + r * 16 + (half ? 8 : 0);
#pragma unroll
    for (int j = 0; j < 4; ++j) {
      int n = nbase + wn * 64 + j * 16 + l16;
      float bv = bias[n];
#pragma unroll
      for (int v = 0; v < 8; ++v) {
        int   m   = mrow0 + v;
        float val = acc[r][j][v] + bv;
        if (MODE == 2) {
          ((float*)out)[(size_t)m * N + n] = val;
        } else {
          int b = m >> 11, s = m & (S_ - 1);     // S_ = 2048
          int h = n >> 7,  d = n & (HD_ - 1);    // HD_ = 128
          size_t off;
          if (MODE == 0) off = ((size_t)(b * NH_ + h) * S_ + s) * HD_ + d;
          else           off = ((size_t)(b * NH_ + h) * HD_ + d) * S_ + s;
          ((bf16_t*)out)[off] = (bf16_t)val;
        }
      }
    }
  }
}

// ---------------- flash attention (causal) ----------------------------------
// grid = (S/64, B*NH), 128 threads = 4 waves; wave owns 16 q rows.
// K step = 32: scores = 2 x (16x16 over 4 K-chunks), online softmax,
// P -> LDS -> A-fragment, 8 PV WMMAs into O[16 x 128].
__global__ void __launch_bounds__(128)
attn_kernel(const bf16_t* __restrict__ Q, const bf16_t* __restrict__ Kt,
            const bf16_t* __restrict__ Vt, bf16_t* __restrict__ ctx) {
  const int bh   = blockIdx.y;            // b*NH + h
  const int b    = bh / NH_, h = bh % NH_;
  const int lane = threadIdx.x & 31;
  const int wave = threadIdx.x >> 5;
  const int q0   = blockIdx.x * 64 + wave * 16;
  const int l16  = lane & 15;
  const int half = (lane < 16) ? 0 : 1;

  const bf16_t* Qh = Q  + (size_t)bh * S_  * HD_;
  const bf16_t* Kh = Kt + (size_t)bh * S_  * HD_;
  const bf16_t* Vh = Vt + (size_t)bh * HD_ * S_;

  // Q fragments: 16 rows x HD, 4 chunks of K=32
  v16bf qf[4];
  {
    const bf16_t* qp = Qh + (size_t)(q0 + l16) * HD_;
#pragma unroll
    for (int t = 0; t < 4; ++t) {
      const bf16_t* p = qp + t * 32 + half * 8;
      qf[t] = bf_cat(*(const v8bf*)p, *(const v8bf*)(p + 16));
    }
  }

  float mrow[8], lrow[8];
  v8f   oacc[8] = {};
#pragma unroll
  for (int v = 0; v < 8; ++v) { mrow[v] = -3.0e38f; lrow[v] = 0.f; }

  __shared__ __align__(64) bf16_t Pl[4][16 * 32];
  bf16_t* Pw = &Pl[wave][0];

  const float scale = 0.088388347648318447f;  // 1/sqrt(128)
  const int   kend  = q0 + 16;                // causal: need k <= q0+15

  for (int kb = 0; kb < kend; kb += 32) {
    // ---- scores: S[16q x 32k] as two 16x16 WMMA chains over d ----
    v8f sacc[2] = {};
#pragma unroll
    for (int j = 0; j < 2; ++j)
#pragma unroll
      for (int t = 0; t < 4; ++t) {
        const bf16_t* p = Kh + (size_t)(kb + j * 16 + l16) * HD_ + t * 32 + half * 8;
        v16bf kf = bf_cat(*(const v8bf*)p, *(const v8bf*)(p + 16));
        sacc[j] = __builtin_amdgcn_wmma_f32_16x16x32_bf16(
            false, qf[t], false, kf, (short)0, sacc[j], false, false);
      }

    // ---- scale + causal mask ----
#pragma unroll
    for (int j = 0; j < 2; ++j) {
      int kcol = kb + j * 16 + l16;
#pragma unroll
      for (int v = 0; v < 8; ++v) {
        float s    = sacc[j][v] * scale;
        int   qrow = q0 + v + (half ? 8 : 0);
        sacc[j][v] = (kcol > qrow) ? -3.0e38f : s;
      }
    }

    // ---- online softmax: row max over 32 cols (16-lane xor reduce) ----
    float alpha[8];
#pragma unroll
    for (int v = 0; v < 8; ++v) {
      float tmax = fmaxf(sacc[0][v], sacc[1][v]);
#pragma unroll
      for (int sh = 8; sh >= 1; sh >>= 1)
        tmax = fmaxf(tmax, __shfl_xor(tmax, sh, 32));
      float mnew = fmaxf(mrow[v], tmax);
      alpha[v]   = __expf(mrow[v] - mnew);
      mrow[v]    = mnew;
    }

    // ---- P = exp(s - m): write bf16 to LDS, accumulate row sums ----
#pragma unroll
    for (int v = 0; v < 8; ++v) {
      float p0 = __expf(sacc[0][v] - mrow[v]);
      float p1 = __expf(sacc[1][v] - mrow[v]);
      int   r  = v + (half ? 8 : 0);
      Pw[r * 32 + l16]      = (bf16_t)p0;
      Pw[r * 32 + 16 + l16] = (bf16_t)p1;
      float ps = p0 + p1;
#pragma unroll
      for (int sh = 8; sh >= 1; sh >>= 1) ps += __shfl_xor(ps, sh, 32);
      lrow[v] = lrow[v] * alpha[v] + ps;
    }

    // ---- rescale running O ----
#pragma unroll
    for (int j = 0; j < 8; ++j)
#pragma unroll
      for (int v = 0; v < 8; ++v) oacc[j][v] *= alpha[v];

    // ---- P (A-frag from LDS) @ Vt (B-frag from global) ----
    const bf16_t* pp  = &Pw[l16 * 32 + half * 8];
    v16bf         paf = bf_cat(*(const v8bf*)pp, *(const v8bf*)(pp + 16));
#pragma unroll
    for (int j = 0; j < 8; ++j) {
      const bf16_t* vp = Vh + (size_t)(j * 16 + l16) * S_ + kb + half * 8;
      v16bf vf = bf_cat(*(const v8bf*)vp, *(const v8bf*)(vp + 16));
      oacc[j] = __builtin_amdgcn_wmma_f32_16x16x32_bf16(
          false, paf, false, vf, (short)0, oacc[j], false, false);
    }
  }

  // ---- normalize + store context [B,S,H] bf16 ----
#pragma unroll
  for (int j = 0; j < 8; ++j)
#pragma unroll
    for (int v = 0; v < 8; ++v) {
      int   q   = q0 + v + (half ? 8 : 0);
      int   d   = j * 16 + l16;
      float val = oacc[j][v] / lrow[v];
      ctx[((size_t)(b * S_ + q)) * H_ + h * HD_ + d] = (bf16_t)val;
    }
}

// ---------------------------------------------------------------------------
extern "C" void kernel_launch(void* const* d_in, const int* in_sizes, int n_in,
                              void* d_out, int out_size, void* d_ws, size_t ws_size,
                              hipStream_t stream) {
  (void)in_sizes; (void)n_in; (void)out_size; (void)ws_size;
  const float* x  = (const float*)d_in[0];
  /* d_in[1] = causal mask: handled analytically */
  const float* Wq = (const float*)d_in[2];
  const float* bq = (const float*)d_in[3];
  const float* Wk = (const float*)d_in[4];
  const float* bk = (const float*)d_in[5];
  const float* Wv = (const float*)d_in[6];
  const float* bv = (const float*)d_in[7];
  const float* Wo = (const float*)d_in[8];
  const float* bo = (const float*)d_in[9];

  char*  ws  = (char*)d_ws;
  size_t off = 0;
  auto alloc = [&](size_t bytes) -> void* {
    void* p = ws + off;
    off += (bytes + 255) & ~(size_t)255;
    return p;
  };
  bf16_t* xb  = (bf16_t*)alloc((size_t)M_ * H_ * 2);
  bf16_t* wqb = (bf16_t*)alloc((size_t)H_ * H_ * 2);
  bf16_t* wkb = (bf16_t*)alloc((size_t)H_ * H_ * 2);
  bf16_t* wvb = (bf16_t*)alloc((size_t)H_ * H_ * 2);
  bf16_t* wob = (bf16_t*)alloc((size_t)H_ * H_ * 2);
  bf16_t* Qb  = (bf16_t*)alloc((size_t)M_ * H_ * 2);
  bf16_t* Kb  = (bf16_t*)alloc((size_t)M_ * H_ * 2);
  bf16_t* Vtb = (bf16_t*)alloc((size_t)M_ * H_ * 2);
  bf16_t* ctx = (bf16_t*)alloc((size_t)M_ * H_ * 2);

  // 1) fp32 -> bf16
  {
    int nx4 = (M_ * H_) / 4, nw4 = (H_ * H_) / 4;
    cvt_f32_bf16_v4<<<(nx4 + 255) / 256, 256, 0, stream>>>((const float4*)x,  (v4bf*)xb,  nx4);
    cvt_f32_bf16_v4<<<(nw4 + 255) / 256, 256, 0, stream>>>((const float4*)Wq, (v4bf*)wqb, nw4);
    cvt_f32_bf16_v4<<<(nw4 + 255) / 256, 256, 0, stream>>>((const float4*)Wk, (v4bf*)wkb, nw4);
    cvt_f32_bf16_v4<<<(nw4 + 255) / 256, 256, 0, stream>>>((const float4*)Wv, (v4bf*)wvb, nw4);
    cvt_f32_bf16_v4<<<(nw4 + 255) / 256, 256, 0, stream>>>((const float4*)Wo, (v4bf*)wob, nw4);
  }

  // 2) projections
  dim3 ggrid(H_ / 128, M_ / 128);  // (16, 32)
  gemm_bf16_nt<0><<<ggrid, 256, 0, stream>>>(xb, wqb, bq, (void*)Qb,  H_, H_);
  gemm_bf16_nt<0><<<ggrid, 256, 0, stream>>>(xb, wkb, bk, (void*)Kb,  H_, H_);
  gemm_bf16_nt<1><<<ggrid, 256, 0, stream>>>(xb, wvb, bv, (void*)Vtb, H_, H_);

  // 3) causal flash attention
  attn_kernel<<<dim3(S_ / 64, B_ * NH_), 128, 0, stream>>>(Qb, Kb, Vtb, ctx);

  // 4) output projection -> fp32 d_out
  gemm_bf16_nt<2><<<ggrid, 256, 0, stream>>>(ctx, wob, bo, d_out, H_, H_);
}
